// MambaBlock_57853209477741
// MI455X (gfx1250) — compile-verified, matrix operand order checked by reading, not run on hardware
//
#include <hip/hip_runtime.h>
#include <math.h>

// ---------------------------------------------------------------------------
// CDNA5 (gfx1250) Mamba block.
// bf16 WMMA GEMMs with one-time operand packing (inner loops = b128 loads +
// v_wmma only), fused epilogues, LDS-broadcast chunked selective scan.
// ---------------------------------------------------------------------------

typedef __bf16 bf16_t;
typedef bf16_t v16bf __attribute__((ext_vector_type(16)));
typedef bf16_t v8bf  __attribute__((ext_vector_type(8)));
typedef bf16_t v4bf  __attribute__((ext_vector_type(4)));
typedef float  v8f   __attribute__((ext_vector_type(8)));

#define WMMA_BF16(A_, B_, C_) \
  __builtin_amdgcn_wmma_f32_16x16x32_bf16(false, (A_), false, (B_), (short)0, (C_), false, false)

constexpr int BATCH = 4;
constexpr int LSEQ  = 2048;
constexpr int DIM   = 1024;
constexpr int HID   = 2048;
constexpr int DST   = 16;
constexpr int DTR   = 64;
constexpr int XPJ   = DTR + 2 * DST;   // 96
constexpr int MROW  = BATCH * LSEQ;    // 8192

__device__ __forceinline__ bf16_t to_bf16(float f) {
  union { float f; unsigned u; } cv; cv.f = f;
  const unsigned u = cv.u;
  const unsigned r = (u + 0x7FFFu + ((u >> 16) & 1u)) >> 16;  // RNE
  const unsigned short s = (unsigned short)r;
  bf16_t h;
  __builtin_memcpy(&h, &s, sizeof(h));
  return h;
}

__device__ __forceinline__ float silu_f(float v) {
  return v / (1.0f + __expf(-v));
}

// Per-lane row pointer for a 16x32 bf16 WMMA operand fragment, row-major src.
// (Also correct for the B operand when the weight is row-major [N][K].)
__device__ __forceinline__ const bf16_t* frag_row_ptr(const bf16_t* __restrict__ base,
                                                      int ld, int row0) {
  const int lane = threadIdx.x & 31;
  const int row  = row0 + (lane & 15);
  const int khi  = (lane >> 4) << 3;  // 0 or 8
  return base + (size_t)row * (size_t)ld + khi;
}

// Fragment load at K offset k0: two contiguous 16-byte chunks per lane.
//   elements 0..7  <- K = khi+0..7     (VGPRs 0-3)
//   elements 8..15 <- K = khi+16..23   (VGPRs 4-7)
__device__ __forceinline__ v16bf load_frag(const bf16_t* __restrict__ rp, int k0) {
  const v8bf lo = *reinterpret_cast<const v8bf*>(rp + k0);
  const v8bf hi = *reinterpret_cast<const v8bf*>(rp + k0 + 16);
  return __builtin_shufflevector(lo, hi, 0, 1, 2, 3, 4, 5, 6, 7,
                                         8, 9, 10, 11, 12, 13, 14, 15);
}

// 16x16 f32 C/D tile store: row = m0 + r + 8*(lane>=16), col = n0 + (lane&15)
__device__ __forceinline__ void store_tile(float* __restrict__ out, int ld,
                                           int m0, int n0, const v8f& acc) {
  const int lane = threadIdx.x & 31;
  const int col  = n0 + (lane & 15);
  const int r0   = m0 + ((lane >> 4) << 3);
#pragma unroll
  for (int r = 0; r < 8; ++r)
    out[(size_t)(r0 + r) * (size_t)ld + col] = acc[r];
}

// ---------------------------------------------------------------------------
// Kernel 0: fp32 -> bf16 pack (4 elements/thread, b128 in, b64 out)
// ---------------------------------------------------------------------------
__global__ void __launch_bounds__(256)
mamba_pack_bf16(const float* __restrict__ in, bf16_t* __restrict__ out, int n4) {
  const int i = blockIdx.x * 256 + threadIdx.x;
  if (i >= n4) return;
  const float4 v = reinterpret_cast<const float4*>(in)[i];
  v4bf o;
  o[0] = to_bf16(v.x); o[1] = to_bf16(v.y);
  o[2] = to_bf16(v.z); o[3] = to_bf16(v.w);
  reinterpret_cast<v4bf*>(out)[i] = o;
}

// ---------------------------------------------------------------------------
// Kernel 1: xz = x @ W_in^T, split into u / z halves. M=8192 K=1024 N=4096.
// Block = 8 waves striding N; each wave owns a 32x32 tile.
// ---------------------------------------------------------------------------
__global__ void __launch_bounds__(256)
mamba_gemm_xz(const bf16_t* __restrict__ xb, const bf16_t* __restrict__ Wb,
              float* __restrict__ u, float* __restrict__ z) {
  const int wave = threadIdx.x >> 5;
  const int m0 = blockIdx.x * 32;
  const int n0 = blockIdx.y * 256 + wave * 32;
  const bf16_t* ra0 = frag_row_ptr(xb, DIM, m0);
  const bf16_t* ra1 = frag_row_ptr(xb, DIM, m0 + 16);
  const bf16_t* rb0 = frag_row_ptr(Wb, DIM, n0);
  const bf16_t* rb1 = frag_row_ptr(Wb, DIM, n0 + 16);
  v8f a00 = {}, a01 = {}, a10 = {}, a11 = {};
  for (int k0 = 0; k0 < DIM; k0 += 32) {
    const v16bf A0 = load_frag(ra0, k0);
    const v16bf A1 = load_frag(ra1, k0);
    const v16bf B0 = load_frag(rb0, k0);
    const v16bf B1 = load_frag(rb1, k0);
    a00 = WMMA_BF16(A0, B0, a00);
    a01 = WMMA_BF16(A0, B1, a01);
    a10 = WMMA_BF16(A1, B0, a10);
    a11 = WMMA_BF16(A1, B1, a11);
  }
  const bool isU = (n0 < HID);
  float* dst = isU ? u : z;
  const int nc = isU ? n0 : (n0 - HID);
  store_tile(dst, HID, m0,      nc,      a00);
  store_tile(dst, HID, m0,      nc + 16, a01);
  store_tile(dst, HID, m0 + 16, nc,      a10);
  store_tile(dst, HID, m0 + 16, nc + 16, a11);
}

// ---------------------------------------------------------------------------
// Kernel 2: depthwise conv (w=3, pad 1, per batch) + bias + SiLU.
// Emits f32 (for scan) and bf16 (for x_dbl GEMM) copies.
// ---------------------------------------------------------------------------
__global__ void __launch_bounds__(256)
mamba_conv_silu(const float* __restrict__ u, const float* __restrict__ cw,
                const float* __restrict__ cb, float* __restrict__ ucf,
                bf16_t* __restrict__ ucb) {
  const int idx = blockIdx.x * 256 + threadIdx.x;   // over MROW*HID
  const int h  = idx & (HID - 1);
  const int bl = idx >> 11;                         // HID == 2048
  const int l  = bl & (LSEQ - 1);
  const float w0 = cw[h * 3 + 0], w1 = cw[h * 3 + 1], w2 = cw[h * 3 + 2];
  float acc = cb[h] + w1 * u[idx];
  if (l > 0)        acc += w0 * u[idx - HID];
  if (l < LSEQ - 1) acc += w2 * u[idx + HID];
  const float s = silu_f(acc);
  ucf[idx] = s;
  ucb[idx] = to_bf16(s);
}

// ---------------------------------------------------------------------------
// Kernel 3: x_dbl = uc @ W_xproj^T. M=8192 K=2048 N=96.
// Stores f32 x_dbl (scan reads B/C) and a bf16 copy of the dtp slice (cols<64).
// ---------------------------------------------------------------------------
__device__ __forceinline__ void store_tile_xdbl(float* __restrict__ xd,
                                                bf16_t* __restrict__ dtp,
                                                int m0, int n0, const v8f& acc) {
  const int lane = threadIdx.x & 31;
  const int col  = n0 + (lane & 15);
  const int r0   = m0 + ((lane >> 4) << 3);
#pragma unroll
  for (int r = 0; r < 8; ++r) {
    const float v = acc[r];
    xd[(size_t)(r0 + r) * XPJ + col] = v;
    if (n0 < DTR) dtp[(size_t)(r0 + r) * DTR + col] = to_bf16(v);
  }
}

__global__ void __launch_bounds__(256)
mamba_gemm_xdbl(const bf16_t* __restrict__ ucb, const bf16_t* __restrict__ Wb,
                float* __restrict__ xdbl, bf16_t* __restrict__ dtp) {
  const int wave = threadIdx.x >> 5;
  const int m0 = blockIdx.x * 256 + wave * 32;
  const int n0 = blockIdx.y * 32;
  const bf16_t* ra0 = frag_row_ptr(ucb, HID, m0);
  const bf16_t* ra1 = frag_row_ptr(ucb, HID, m0 + 16);
  const bf16_t* rb0 = frag_row_ptr(Wb, HID, n0);
  const bf16_t* rb1 = frag_row_ptr(Wb, HID, n0 + 16);
  v8f a00 = {}, a01 = {}, a10 = {}, a11 = {};
  for (int k0 = 0; k0 < HID; k0 += 32) {
    const v16bf A0 = load_frag(ra0, k0);
    const v16bf A1 = load_frag(ra1, k0);
    const v16bf B0 = load_frag(rb0, k0);
    const v16bf B1 = load_frag(rb1, k0);
    a00 = WMMA_BF16(A0, B0, a00);
    a01 = WMMA_BF16(A0, B1, a01);
    a10 = WMMA_BF16(A1, B0, a10);
    a11 = WMMA_BF16(A1, B1, a11);
  }
  store_tile_xdbl(xdbl, dtp, m0,      n0,      a00);
  store_tile_xdbl(xdbl, dtp, m0,      n0 + 16, a01);
  store_tile_xdbl(xdbl, dtp, m0 + 16, n0,      a10);
  store_tile_xdbl(xdbl, dtp, m0 + 16, n0 + 16, a11);
}

// ---------------------------------------------------------------------------
// Kernel 4: dt = softplus(dtp @ W_dt^T + b_dt). M=8192 K=64 N=2048.
// ---------------------------------------------------------------------------
__device__ __forceinline__ void store_tile_softplus(float* __restrict__ out,
                                                    const float* __restrict__ b_dt,
                                                    int m0, int n0, const v8f& acc) {
  const int lane = threadIdx.x & 31;
  const int col  = n0 + (lane & 15);
  const int r0   = m0 + ((lane >> 4) << 3);
  const float bb = b_dt[col];
#pragma unroll
  for (int r = 0; r < 8; ++r) {
    float v = acc[r] + bb;
    v = (v > 20.0f) ? v : log1pf(__expf(v));
    out[(size_t)(r0 + r) * HID + col] = v;
  }
}

__global__ void __launch_bounds__(256)
mamba_gemm_dt(const bf16_t* __restrict__ dtp, const bf16_t* __restrict__ Wb,
              const float* __restrict__ b_dt, float* __restrict__ dt) {
  const int wave = threadIdx.x >> 5;
  const int m0 = blockIdx.x * 32;
  const int n0 = blockIdx.y * 256 + wave * 32;
  const bf16_t* ra0 = frag_row_ptr(dtp, DTR, m0);
  const bf16_t* ra1 = frag_row_ptr(dtp, DTR, m0 + 16);
  const bf16_t* rb0 = frag_row_ptr(Wb, DTR, n0);
  const bf16_t* rb1 = frag_row_ptr(Wb, DTR, n0 + 16);
  v8f a00 = {}, a01 = {}, a10 = {}, a11 = {};
#pragma unroll
  for (int k0 = 0; k0 < DTR; k0 += 32) {
    const v16bf A0 = load_frag(ra0, k0);
    const v16bf A1 = load_frag(ra1, k0);
    const v16bf B0 = load_frag(rb0, k0);
    const v16bf B1 = load_frag(rb1, k0);
    a00 = WMMA_BF16(A0, B0, a00);
    a01 = WMMA_BF16(A0, B1, a01);
    a10 = WMMA_BF16(A1, B0, a10);
    a11 = WMMA_BF16(A1, B1, a11);
  }
  store_tile_softplus(dt, b_dt, m0,      n0,      a00);
  store_tile_softplus(dt, b_dt, m0,      n0 + 16, a01);
  store_tile_softplus(dt, b_dt, m0 + 16, n0,      a10);
  store_tile_softplus(dt, b_dt, m0 + 16, n0 + 16, a11);
}

// ---------------------------------------------------------------------------
// Kernel 5: selective scan (sequential over L). One thread per (b, h) channel;
// 16-state recurrence in registers. B/C staged in LDS in chunks of 16 steps.
// Fuses +D*u and *silu(z); emits y directly as bf16 for the output GEMM.
// ---------------------------------------------------------------------------
constexpr int SCAN_CH = 16;

__global__ void __launch_bounds__(256)
mamba_scan(const float* __restrict__ uc, const float* __restrict__ dt,
           const float* __restrict__ xdbl, const float* __restrict__ A_log,
           const float* __restrict__ D, const float* __restrict__ z,
           bf16_t* __restrict__ y) {
  const int h = blockIdx.x * 256 + threadIdx.x;   // hidden channel
  const int b = blockIdx.y;                       // batch
  __shared__ float sB[SCAN_CH][DST];
  __shared__ float sC[SCAN_CH][DST];
  float Aval[DST];
#pragma unroll
  for (int n = 0; n < DST; ++n) Aval[n] = -__expf(A_log[n]);
  const float Dh = D[h];
  float st[DST];
#pragma unroll
  for (int n = 0; n < DST; ++n) st[n] = 0.0f;
  const size_t baseBL = (size_t)b * LSEQ;
  for (int l0 = 0; l0 < LSEQ; l0 += SCAN_CH) {
    // cooperative stage of SCAN_CH x 32 B/C values
    for (int e = threadIdx.x; e < SCAN_CH * 32; e += 256) {
      const int ll = e >> 5, c = e & 31;
      const float v = xdbl[(baseBL + l0 + ll) * XPJ + DTR + c];
      if (c < DST) sB[ll][c] = v;
      else         sC[ll][c - DST] = v;
    }
    __syncthreads();
#pragma unroll 4
    for (int j = 0; j < SCAN_CH; ++j) {
      const size_t idx = (baseBL + l0 + j) * HID + h;
      const float u_t  = uc[idx];
      const float dt_t = dt[idx];
      float yv = 0.0f;
#pragma unroll
      for (int n = 0; n < DST; ++n) {
        const float dA = __expf(dt_t * Aval[n]);
        st[n] = dA * st[n] + sB[j][n] * u_t;
        yv += sC[j][n] * st[n];
      }
      y[idx] = to_bf16((yv + Dh * u_t) * silu_f(z[idx]));
    }
    __syncthreads();
  }
}

// ---------------------------------------------------------------------------
// Kernel 6: out = x + y @ W_out^T. M=8192 K=2048 N=1024. Residual fused.
// ---------------------------------------------------------------------------
__device__ __forceinline__ void store_tile_residual(float* __restrict__ out,
                                                    const float* __restrict__ x,
                                                    int m0, int n0, const v8f& acc) {
  const int lane = threadIdx.x & 31;
  const int col  = n0 + (lane & 15);
  const int r0   = m0 + ((lane >> 4) << 3);
#pragma unroll
  for (int r = 0; r < 8; ++r) {
    const size_t off = (size_t)(r0 + r) * DIM + col;
    out[off] = x[off] + acc[r];
  }
}

__global__ void __launch_bounds__(256)
mamba_gemm_out(const bf16_t* __restrict__ yb, const bf16_t* __restrict__ Wb,
               const float* __restrict__ x, float* __restrict__ out) {
  const int wave = threadIdx.x >> 5;
  const int m0 = blockIdx.x * 32;
  const int n0 = blockIdx.y * 256 + wave * 32;
  const bf16_t* ra0 = frag_row_ptr(yb, HID, m0);
  const bf16_t* ra1 = frag_row_ptr(yb, HID, m0 + 16);
  const bf16_t* rb0 = frag_row_ptr(Wb, HID, n0);
  const bf16_t* rb1 = frag_row_ptr(Wb, HID, n0 + 16);
  v8f a00 = {}, a01 = {}, a10 = {}, a11 = {};
  for (int k0 = 0; k0 < HID; k0 += 32) {
    const v16bf A0 = load_frag(ra0, k0);
    const v16bf A1 = load_frag(ra1, k0);
    const v16bf B0 = load_frag(rb0, k0);
    const v16bf B1 = load_frag(rb1, k0);
    a00 = WMMA_BF16(A0, B0, a00);
    a01 = WMMA_BF16(A0, B1, a01);
    a10 = WMMA_BF16(A1, B0, a10);
    a11 = WMMA_BF16(A1, B1, a11);
  }
  store_tile_residual(out, x, m0,      n0,      a00);
  store_tile_residual(out, x, m0,      n0 + 16, a01);
  store_tile_residual(out, x, m0 + 16, n0,      a10);
  store_tile_residual(out, x, m0 + 16, n0 + 16, a11);
}

// ---------------------------------------------------------------------------
extern "C" void kernel_launch(void* const* d_in, const int* in_sizes, int n_in,
                              void* d_out, int out_size, void* d_ws, size_t ws_size,
                              hipStream_t stream) {
  const float* x       = (const float*)d_in[0];
  const float* W_in    = (const float*)d_in[1];
  const float* conv_w  = (const float*)d_in[2];
  const float* conv_b  = (const float*)d_in[3];
  const float* W_xproj = (const float*)d_in[4];
  const float* W_dt    = (const float*)d_in[5];
  const float* b_dt    = (const float*)d_in[6];
  const float* A_log   = (const float*)d_in[7];
  const float* D       = (const float*)d_in[8];
  const float* W_out   = (const float*)d_in[9];
  float* out = (float*)d_out;

  // ---- workspace carve-up (256B aligned) ----
  char* base = (char*)d_ws;
  auto alloc = [&](size_t bytes) -> char* {
    char* p = base;
    base += (bytes + 255) & ~(size_t)255;
    return p;
  };
  float*  uBuf   = (float*)alloc((size_t)MROW * HID * 4);      // also reused as dt
  float*  zBuf   = (float*)alloc((size_t)MROW * HID * 4);
  float*  ucf    = (float*)alloc((size_t)MROW * HID * 4);
  float*  xdbl   = (float*)alloc((size_t)MROW * XPJ * 4);
  bf16_t* xbf    = (bf16_t*)alloc((size_t)MROW * DIM * 2);
  bf16_t* wInBf  = (bf16_t*)alloc((size_t)2 * HID * DIM * 2);
  bf16_t* ucb    = (bf16_t*)alloc((size_t)MROW * HID * 2);
  bf16_t* wXpBf  = (bf16_t*)alloc((size_t)XPJ * HID * 2);
  bf16_t* dtpBf  = (bf16_t*)alloc((size_t)MROW * DTR * 2);
  bf16_t* wDtBf  = (bf16_t*)alloc((size_t)HID * DTR * 2);
  bf16_t* ybf    = (bf16_t*)alloc((size_t)MROW * HID * 2);
  bf16_t* wOutBf = (bf16_t*)alloc((size_t)DIM * HID * 2);
  float*  dtBuf  = uBuf;   // u is dead after the conv kernel

  auto pack = [&](const float* src, bf16_t* dst, size_t n) {
    const int n4 = (int)(n / 4);
    mamba_pack_bf16<<<(n4 + 255) / 256, 256, 0, stream>>>(src, dst, n4);
  };
  pack(x,       xbf,    (size_t)MROW * DIM);
  pack(W_in,    wInBf,  (size_t)2 * HID * DIM);
  pack(W_xproj, wXpBf,  (size_t)XPJ * HID);
  pack(W_dt,    wDtBf,  (size_t)HID * DTR);
  pack(W_out,   wOutBf, (size_t)DIM * HID);

  mamba_gemm_xz  <<<dim3(MROW / 32, (2 * HID) / 256), 256, 0, stream>>>(xbf, wInBf, uBuf, zBuf);
  mamba_conv_silu<<<(MROW * HID) / 256, 256, 0, stream>>>(uBuf, conv_w, conv_b, ucf, ucb);
  mamba_gemm_xdbl<<<dim3(MROW / 256, XPJ / 32), 256, 0, stream>>>(ucb, wXpBf, xdbl, dtpBf);
  mamba_gemm_dt  <<<dim3(MROW / 32, HID / 256), 256, 0, stream>>>(dtpBf, wDtBf, b_dt, dtBuf);
  mamba_scan     <<<dim3(HID / 256, BATCH), 256, 0, stream>>>(ucf, dtBuf, xdbl, A_log, D, zBuf, ybf);
  mamba_gemm_out <<<dim3(MROW / 32, DIM / 256), 256, 0, stream>>>(ybf, wOutBf, x, out);
}
